// EncoderDecoder_2851858284768
// MI455X (gfx1250) — compile-verified, matrix operand order checked by reading, not run on hardware
//
#include <hip/hip_runtime.h>

#define VOCAB 50257
#define EDIM  512
#define HDIM  1024
#define ADIM  64
#define SLEN  64
#define TLEN  64
#define G4    (4*HDIM)   // 4096

typedef __attribute__((ext_vector_type(16))) __bf16 v16bf;
typedef __attribute__((ext_vector_type(8)))  float  v8f;

// Accurate RNE conversion (used on small, one-shot data).
__device__ __forceinline__ __bf16 f2bf(float x) {
    unsigned u = __builtin_bit_cast(unsigned, x);
    unsigned r = u + 0x7FFFu + ((u >> 16) & 1u);
    unsigned short s = (unsigned short)(r >> 16);
    return __builtin_bit_cast(__bf16, s);
}

// Fast packed conversion for the streamed GEMM operand:
// one v_add per element + one v_perm_b32 per pair (high 16 bits of each word).
__device__ __forceinline__ unsigned pack_bf16(float lo, float hi) {
    unsigned ulo = __builtin_bit_cast(unsigned, lo) + 0x8000u;   // round-half-up
    unsigned uhi = __builtin_bit_cast(unsigned, hi) + 0x8000u;
    return __builtin_amdgcn_perm(uhi, ulo, 0x07060302u);         // {hi[31:16],lo[31:16]}
}

__device__ __forceinline__ float sigmoidf(float x) {
    return 1.0f / (1.0f + expf(-x));
}

// ---------------------------------------------------------------------------
// Gather embeddings -> bf16 activation matrices; zero initial LSTM state.
// Aenc: [64,512] bf16   Adec: [64,576] bf16 (emb || features)
// ---------------------------------------------------------------------------
__global__ void embed_gather(const int* __restrict__ enc_ids,
                             const int* __restrict__ dec_ids,
                             const float* __restrict__ emb,
                             const float* __restrict__ feats,
                             __bf16* __restrict__ Aenc,
                             __bf16* __restrict__ Adec,
                             float* __restrict__ h, float* __restrict__ c) {
    int s = blockIdx.x, tid = threadIdx.x;
    const float* er = emb + (size_t)enc_ids[s] * EDIM;
    for (int i = tid; i < EDIM; i += 256) Aenc[s * EDIM + i] = f2bf(er[i]);
    const float* dr = emb + (size_t)dec_ids[s] * EDIM;
    for (int i = tid; i < EDIM; i += 256) Adec[s * (EDIM + ADIM) + i] = f2bf(dr[i]);
    for (int i = tid; i < ADIM; i += 256) Adec[s * (EDIM + ADIM) + EDIM + i] = f2bf(feats[i]);
    if (s == 0)
        for (int i = tid; i < HDIM; i += 256) { h[i] = 0.f; c[i] = 0.f; }
}

// ---------------------------------------------------------------------------
// WMMA GEMM: C[64,N] = A[64,K](bf16) @ B[N,K](f32, converted to bf16)^T + bias
// Block: 256 thr = 8 waves. Each wave: unique 16-col N tile, loops 4 M tiles.
// 4x v_wmma_f32_16x16x32_bf16 per K-step of 32.
// ---------------------------------------------------------------------------
__global__ __launch_bounds__(256) void gemm_bf16_tn(
        const __bf16* __restrict__ A, int lda,
        const float* __restrict__ B, int N, int K,
        const float* __restrict__ bias0, const float* __restrict__ bias1,
        float* __restrict__ C, int ldc, int relu) {
    int wave = threadIdx.x >> 5, lane = threadIdx.x & 31;
    int half = lane >> 4, l = lane & 15;
    int nTile = blockIdx.x * 128 + wave * 16;
    int n  = nTile + l;
    int nB = (n < N) ? n : (N - 1);                 // clamp: avoid divergence pre-WMMA
    const float* rowB = B + (size_t)nB * K;

    v8f acc[4] = {};

    for (int kk = 0; kk < K; kk += 32) {
        int c0 = kk + half * 8;                     // lanes 0-15: K {0..7,16..23}
        int c1 = c0 + 16;                           // lanes 16-31: K {8..15,24..31}
        __builtin_prefetch((const void*)(rowB + kk + 128), 0, 0);

        float4 b0 = *(const float4*)(rowB + c0);
        float4 b1 = *(const float4*)(rowB + c0 + 4);
        float4 b2 = *(const float4*)(rowB + c1);
        float4 b3 = *(const float4*)(rowB + c1 + 4);
        union { v16bf v; unsigned u[8]; } b;
        b.u[0] = pack_bf16(b0.x, b0.y);
        b.u[1] = pack_bf16(b0.z, b0.w);
        b.u[2] = pack_bf16(b1.x, b1.y);
        b.u[3] = pack_bf16(b1.z, b1.w);
        b.u[4] = pack_bf16(b2.x, b2.y);
        b.u[5] = pack_bf16(b2.z, b2.w);
        b.u[6] = pack_bf16(b3.x, b3.y);
        b.u[7] = pack_bf16(b3.z, b3.w);

        #pragma unroll
        for (int mt = 0; mt < 4; ++mt) {
            const __bf16* rA = A + (size_t)(mt * 16 + l) * lda;
            union { v16bf v; uint4 q[2]; } a;
            a.q[0] = *(const uint4*)(rA + c0);
            a.q[1] = *(const uint4*)(rA + c1);
            acc[mt] = __builtin_amdgcn_wmma_f32_16x16x32_bf16(
                false, a.v, false, b.v, (short)0, acc[mt], false, false);
        }
    }

    if (n < N) {
        float bb = (bias0 ? bias0[n] : 0.f) + (bias1 ? bias1[n] : 0.f);
        #pragma unroll
        for (int mt = 0; mt < 4; ++mt) {
            #pragma unroll
            for (int j = 0; j < 8; ++j) {
                int m = mt * 16 + j + half * 8;     // D layout: VGPR j, lane half
                float v = acc[mt][j] + bb;
                if (relu) v = fmaxf(v, 0.f);
                C[(size_t)m * ldc + n] = v;
            }
        }
    }
}

// ---------------------------------------------------------------------------
// Recurrent matvec: g[4096] = Gpre[4096] + Whh[4096,1024] @ h[1024]
// One wave per output row; wave32 shuffle reduction. Whh is L2-resident.
// ---------------------------------------------------------------------------
__global__ __launch_bounds__(256) void lstm_matvec(
        const float* __restrict__ W, const float* __restrict__ h,
        const float* __restrict__ gpre, float* __restrict__ g) {
    int wv = threadIdx.x >> 5, lane = threadIdx.x & 31;
    int row = blockIdx.x * 8 + wv;                  // 512 blocks -> 4096 rows
    const float* wr = W + (size_t)row * HDIM;
    float acc = 0.f;
    #pragma unroll
    for (int k0 = 0; k0 < HDIM; k0 += 128) {
        int k = k0 + lane * 4;
        float4 w4 = *(const float4*)(wr + k);
        float4 h4 = *(const float4*)(h + k);
        acc += w4.x * h4.x + w4.y * h4.y + w4.z * h4.z + w4.w * h4.w;
    }
    #pragma unroll
    for (int off = 16; off; off >>= 1) acc += __shfl_down(acc, off, 32);
    if (lane == 0) g[row] = gpre[row] + acc;
}

// ---------------------------------------------------------------------------
// LSTM cell elementwise update (PyTorch gate order i,f,g,o).
// ---------------------------------------------------------------------------
__global__ void lstm_cell(const float* __restrict__ g, float* __restrict__ h,
                          float* __restrict__ c, __bf16* __restrict__ hs_bf) {
    int i = blockIdx.x * blockDim.x + threadIdx.x;
    if (i >= HDIM) return;
    float ig = sigmoidf(g[i]);
    float fg = sigmoidf(g[HDIM + i]);
    float gg = tanhf(g[2 * HDIM + i]);
    float og = sigmoidf(g[3 * HDIM + i]);
    float cc = fg * c[i] + ig * gg;
    float hh = og * tanhf(cc);
    c[i] = cc; h[i] = hh;
    if (hs_bf) hs_bf[i] = f2bf(hh);
}

// ---------------------------------------------------------------------------
// In-place row-wise log_softmax over V (input already ReLU'd by GEMM epilogue).
// ---------------------------------------------------------------------------
__global__ __launch_bounds__(256) void logsoftmax_inplace(float* __restrict__ out, int n) {
    float* row = out + (size_t)blockIdx.x * n;
    __shared__ float red[8];
    int tid = threadIdx.x, lane = tid & 31, wv = tid >> 5;

    float m = -1e30f;
    for (int i = tid; i < n; i += 256) m = fmaxf(m, row[i]);
    #pragma unroll
    for (int off = 16; off; off >>= 1) m = fmaxf(m, __shfl_down(m, off, 32));
    if (lane == 0) red[wv] = m;
    __syncthreads();
    float bmax = red[0];
    #pragma unroll
    for (int i = 1; i < 8; ++i) bmax = fmaxf(bmax, red[i]);
    __syncthreads();

    float s = 0.f;
    for (int i = tid; i < n; i += 256) s += expf(row[i] - bmax);
    #pragma unroll
    for (int off = 16; off; off >>= 1) s += __shfl_down(s, off, 32);
    if (lane == 0) red[wv] = s;
    __syncthreads();
    float bsum = 0.f;
    #pragma unroll
    for (int i = 0; i < 8; ++i) bsum += red[i];
    float lg = logf(bsum);
    for (int i = tid; i < n; i += 256) row[i] = row[i] - bmax - lg;
}

// ---------------------------------------------------------------------------
extern "C" void kernel_launch(void* const* d_in, const int* in_sizes, int n_in,
                              void* d_out, int out_size, void* d_ws, size_t ws_size,
                              hipStream_t stream) {
    (void)in_sizes; (void)n_in; (void)out_size; (void)ws_size;
    const int*   in_seq  = (const int*)  d_in[0];
    const float* feats   = (const float*)d_in[1];
    const int*   dec_seq = (const int*)  d_in[2];
    const float* emb     = (const float*)d_in[3];
    const float* eWih    = (const float*)d_in[4];
    const float* eWhh    = (const float*)d_in[5];
    const float* ebih    = (const float*)d_in[6];
    const float* ebhh    = (const float*)d_in[7];
    const float* dWih    = (const float*)d_in[8];
    const float* dWhh    = (const float*)d_in[9];
    const float* dbih    = (const float*)d_in[10];
    const float* dbhh    = (const float*)d_in[11];
    const float* linW    = (const float*)d_in[12];
    const float* linb    = (const float*)d_in[13];
    float* out = (float*)d_out;

    char* ws = (char*)d_ws;
    size_t off = 0;
    auto alloc = [&](size_t bytes) -> void* {
        void* p = ws + off;
        off += (bytes + 255) & ~(size_t)255;
        return p;
    };
    __bf16* Aenc = (__bf16*)alloc((size_t)SLEN * EDIM * 2);
    __bf16* Adec = (__bf16*)alloc((size_t)TLEN * (EDIM + ADIM) * 2);
    __bf16* hsbf = (__bf16*)alloc((size_t)TLEN * HDIM * 2);
    float*  Gen  = (float*) alloc((size_t)SLEN * G4 * 4);
    float*  Gde  = (float*) alloc((size_t)TLEN * G4 * 4);
    float*  g    = (float*) alloc((size_t)G4 * 4);
    float*  h    = (float*) alloc((size_t)HDIM * 4);
    float*  c    = (float*) alloc((size_t)HDIM * 4);

    // 1) embed + init state
    embed_gather<<<SLEN, 256, 0, stream>>>(in_seq, dec_seq, emb, feats, Aenc, Adec, h, c);

    // 2) batched input projections (WMMA): Gpre = x @ Wih^T + bih + bhh
    gemm_bf16_tn<<<G4 / 128, 256, 0, stream>>>(Aenc, EDIM, eWih, G4, EDIM,
                                               ebih, ebhh, Gen, G4, 0);
    gemm_bf16_tn<<<G4 / 128, 256, 0, stream>>>(Adec, EDIM + ADIM, dWih, G4, EDIM + ADIM,
                                               dbih, dbhh, Gde, G4, 0);

    // 3) encoder recurrence (sequential, Whh L2-resident)
    for (int t = 0; t < SLEN; ++t) {
        lstm_matvec<<<G4 / 8, 256, 0, stream>>>(eWhh, h, Gen + (size_t)t * G4, g);
        lstm_cell<<<HDIM / 256, 256, 0, stream>>>(g, h, c, (__bf16*)nullptr);
    }

    // 4) decoder recurrence (teacher forcing), record bf16 hidden states
    for (int t = 0; t < TLEN; ++t) {
        lstm_matvec<<<G4 / 8, 256, 0, stream>>>(dWhh, h, Gde + (size_t)t * G4, g);
        lstm_cell<<<HDIM / 256, 256, 0, stream>>>(g, h, c, hsbf + (size_t)t * HDIM);
    }

    // 5) logits = relu(hs @ lin_W^T + lin_b), streamed 206 MB -> HBM bound
    gemm_bf16_tn<<<(VOCAB + 127) / 128, 256, 0, stream>>>(hsbf, HDIM, linW, VOCAB, HDIM,
                                                          linb, nullptr, out, VOCAB, 1);

    // 6) in-place log_softmax per row
    logsoftmax_inplace<<<TLEN, 256, 0, stream>>>(out, VOCAB);
}
